// Sketch_RNN_91036126806751
// MI455X (gfx1250) — compile-verified, matrix operand order checked by reading, not run on hardware
//
#include <hip/hip_runtime.h>
#include <hip/hip_bf16.h>
#include <math.h>

typedef __attribute__((ext_vector_type(16))) __bf16 bf16x16;
typedef __attribute__((ext_vector_type(8)))  __bf16 bf16x8;
typedef __attribute__((ext_vector_type(8)))  float  f32x8;

#define B_  2048
#define T_  256
#define P_  5
#define Z_  128
#define H_  1024
#define K_  20
#define O_  123
#define G4H 4096   // 4*H

// ---------------------------------------------------------------------------
// CDNA5 async copy helpers (ASYNCcnt path; see CDNA5 ISA)
// ---------------------------------------------------------------------------
__device__ __forceinline__ void async_copy_b128(void* lds_dst, const void* gsrc) {
  unsigned lds = (unsigned)(size_t)((__attribute__((address_space(3))) void*)lds_dst);
  unsigned long long ga = (unsigned long long)(size_t)gsrc;
  asm volatile("global_load_async_to_lds_b128 %0, %1, off"
               :: "v"(lds), "v"(ga) : "memory");
}
__device__ __forceinline__ void wait_async0() {
  asm volatile("s_wait_asynccnt 0x0" ::: "memory");
}

// ---------------------------------------------------------------------------
// bf16 WMMA GEMM:  C[M,N] = A[M,Kd] * W[N,Kd]^T   (f32 accumulate, raw store)
// Block tile 128x128, BK=64 (2 sub-steps of 32), 256 threads (8 waves),
// wave tile 32(M) x 64(N). Double-buffered LDS filled with async copies.
// Requires M%128==0, N%128==0, Kd%64==0 (true for all GEMMs in this model).
// ---------------------------------------------------------------------------
#define BM 128
#define BN 128
#define BK 64
#define LDP (BK + 8)   // padded LDS row stride (bf16 elems)

__global__ __launch_bounds__(256)
void gemm_bf16(const __bf16* __restrict__ A, const __bf16* __restrict__ W,
               float* __restrict__ C, int M, int N, int Kd) {
  __shared__ __bf16 sA[2][BM][LDP];
  __shared__ __bf16 sW[2][BN][LDP];

  const int tid  = threadIdx.x;
  const int wave = tid >> 5;
  const int lane = tid & 31;
  const int m0 = blockIdx.x * BM;
  const int n0 = blockIdx.y * BN;

  // wave tile origin within block
  const int wm = (wave >> 1) * 32;   // 0,32,64,96
  const int wn = (wave & 1) * 64;    // 0,64

  const int lrow  = lane & 15;
  const int khalf = (lane < 16) ? 0 : 8;   // ISA 16-bit A layout: lane half picks K-half

  f32x8 acc[2][4] = {};

  // cooperative async fill of one double-buffer slot:
  // 128x64 bf16 tile = 1024 16B-chunks per matrix, 4 per thread per matrix
  auto issue_tile = [&](int buf, int k0) {
#pragma unroll
    for (int cch = 0; cch < 4; ++cch) {
      int chunk = tid * 4 + cch;           // 0..1023
      int row   = chunk >> 3;              // 0..127
      int col   = (chunk & 7) * 8;         // 0,8,...,56
      async_copy_b128(&sA[buf][row][col], &A[(size_t)(m0 + row) * Kd + k0 + col]);
      async_copy_b128(&sW[buf][row][col], &W[(size_t)(n0 + row) * Kd + k0 + col]);
    }
  };

  const int nk = Kd / BK;
  issue_tile(0, 0);

  for (int i = 0; i < nk; ++i) {
    const int cur = i & 1;
    wait_async0();       // my async writes into buf cur have landed in LDS
    __syncthreads();     // all waves' writes visible; all reads of buf cur^1 retired

    if (i + 1 < nk) issue_tile(cur ^ 1, (i + 1) * BK);

#pragma unroll
    for (int ks = 0; ks < 2; ++ks) {
      const int kb = ks * 32;
      bf16x16 afrag[2], wfrag[4];
#pragma unroll
      for (int ii = 0; ii < 2; ++ii) {
        bf16x8 lo = *(const bf16x8*)&sA[cur][wm + ii * 16 + lrow][kb + khalf];
        bf16x8 hi = *(const bf16x8*)&sA[cur][wm + ii * 16 + lrow][kb + 16 + khalf];
#pragma unroll
        for (int e = 0; e < 8; ++e) { afrag[ii][e] = lo[e]; afrag[ii][8 + e] = hi[e]; }
      }
#pragma unroll
      for (int j = 0; j < 4; ++j) {
        bf16x8 lo = *(const bf16x8*)&sW[cur][wn + j * 16 + lrow][kb + khalf];
        bf16x8 hi = *(const bf16x8*)&sW[cur][wn + j * 16 + lrow][kb + 16 + khalf];
#pragma unroll
        for (int e = 0; e < 8; ++e) { wfrag[j][e] = lo[e]; wfrag[j][8 + e] = hi[e]; }
      }
#pragma unroll
      for (int ii = 0; ii < 2; ++ii)
#pragma unroll
        for (int j = 0; j < 4; ++j)
          acc[ii][j] = __builtin_amdgcn_wmma_f32_16x16x32_bf16(
              false, afrag[ii], false, wfrag[j], (short)0, acc[ii][j], false, false);
    }
  }

  // store: C/D layout — VGPR r holds M=r (lanes 0-15) and M=r+8 (lanes 16-31)
  const int rbase = (lane < 16) ? 0 : 8;
#pragma unroll
  for (int ii = 0; ii < 2; ++ii)
#pragma unroll
    for (int j = 0; j < 4; ++j) {
      int mrow = m0 + wm + ii * 16 + rbase;
      int ncol = n0 + wn + j * 16 + lrow;
#pragma unroll
      for (int r = 0; r < 8; ++r)
        C[(size_t)(mrow + r) * N + ncol] = acc[ii][j][r];
    }
}

// ---------------------------------------------------------------------------
// one-time conversion / setup kernels
// ---------------------------------------------------------------------------
__global__ void cvt_bf16(const float* __restrict__ src, __bf16* __restrict__ dst, int n) {
  int i = blockIdx.x * blockDim.x + threadIdx.x;
  if (i < n) dst[i] = (__bf16)src[i];
}

// w_ih is (4H, P+Z); extract columns [P, P+Z) as bf16 (4H x Z)
__global__ void extract_wihz(const float* __restrict__ w_ih, __bf16* __restrict__ dst) {
  int i = blockIdx.x * blockDim.x + threadIdx.x;   // 4H*Z
  if (i >= G4H * Z_) return;
  int row = i >> 7;          // /Z_
  int col = i & (Z_ - 1);
  dst[i] = (__bf16)w_ih[(size_t)row * (P_ + Z_) + P_ + col];
}

// pad w_dec (O x H) to (128 x H) bf16 with zero rows
__global__ void pad_wdec(const float* __restrict__ w_dec, __bf16* __restrict__ dst) {
  int i = blockIdx.x * blockDim.x + threadIdx.x;   // 128*H
  if (i >= 128 * H_) return;
  int row = i >> 10;
  int col = i & (H_ - 1);
  dst[i] = (row < O_) ? (__bf16)w_dec[(size_t)row * H_ + col] : (__bf16)0.0f;
}

// h = tanh(h0_raw + b_h0), c = 0
__global__ void h0_init(const float* __restrict__ raw, const float* __restrict__ b_h0,
                        __bf16* __restrict__ h, float* __restrict__ c) {
  int i = blockIdx.x * blockDim.x + threadIdx.x;   // B*H
  if (i >= B_ * H_) return;
  int n = i & (H_ - 1);
  h[i] = (__bf16)tanhf(raw[i] + b_h0[n]);
  c[i] = 0.0f;
}

// zero qk sums (T) and the mu/presig tail of d_out (2*Z floats)
__global__ void init_misc(float* __restrict__ qk_sum, float* __restrict__ out_tail) {
  int i = threadIdx.x;                              // 256 threads
  if (i < T_) qk_sum[i] = 0.0f;
  out_tail[i] = 0.0f;                               // 2*Z == 256
}

// ---------------------------------------------------------------------------
// LSTM cell: gates = graw + gbase + b_ih + b_hh + s_t . w_ih[:, :P]
// ---------------------------------------------------------------------------
__device__ __forceinline__ float sigmoidf_(float x) { return 1.0f / (1.0f + expf(-x)); }

__global__ __launch_bounds__(256)
void lstm_cell(const float* __restrict__ graw, const float* __restrict__ gbase,
               const float* __restrict__ b_ih, const float* __restrict__ b_hh,
               const float* __restrict__ s,    const float* __restrict__ w_ih,
               float* __restrict__ c, __bf16* __restrict__ h, int t) {
  int idx = blockIdx.x * blockDim.x + threadIdx.x;   // b*H + hidx
  if (idx >= B_ * H_) return;
  int b    = idx >> 10;
  int hidx = idx & (H_ - 1);

  float sv[P_];
  if (t == 0) {
    sv[0] = 0.f; sv[1] = 0.f; sv[2] = 1.f; sv[3] = 0.f; sv[4] = 0.f;
  } else {
    const float* sp = s + ((size_t)b * T_ + (t - 1)) * P_;
#pragma unroll
    for (int p = 0; p < P_; ++p) sv[p] = sp[p];
  }

  float g[4];
#pragma unroll
  for (int q = 0; q < 4; ++q) {
    int n = q * H_ + hidx;
    float v = graw[(size_t)b * G4H + n] + gbase[(size_t)b * G4H + n] + b_ih[n] + b_hh[n];
    const float* w = w_ih + (size_t)n * (P_ + Z_);
#pragma unroll
    for (int p = 0; p < P_; ++p) v += sv[p] * w[p];
    g[q] = v;
  }

  float cn = sigmoidf_(g[1]) * c[idx] + sigmoidf_(g[0]) * tanhf(g[2]);
  float hn = sigmoidf_(g[3]) * tanhf(cn);
  c[idx] = cn;
  h[idx] = (__bf16)hn;
}

// ---------------------------------------------------------------------------
// output heads: pi softmax over K, mean copy, exp std, tanh cor, exp qk + sum
// ---------------------------------------------------------------------------
__global__ __launch_bounds__(128)
void decode_out(const float* __restrict__ yraw, const float* __restrict__ b_dec,
                float* __restrict__ out, float* __restrict__ qk_sum, int t) {
  __shared__ float sy[128];
  int b = blockIdx.x;
  int j = threadIdx.x;
  float v = yraw[(size_t)b * 128 + j] + ((j < O_) ? b_dec[j] : 0.0f);
  sy[j] = v;
  __syncthreads();

  float* orow = out + ((size_t)b * T_ + t) * O_;
  if (j < 6 * K_) {
    int k = j / 6, r = j - 6 * k;
    float res;
    if (r == 0) {
      float m = -3.0e38f;
#pragma unroll 4
      for (int kk = 0; kk < K_; ++kk) m = fmaxf(m, sy[6 * kk]);
      float sum = 0.0f;
#pragma unroll 4
      for (int kk = 0; kk < K_; ++kk) sum += expf(sy[6 * kk] - m);
      res = expf(v - m) / sum;
    } else if (r < 3) {
      res = v;
    } else if (r < 5) {
      res = expf(v);
    } else {
      res = tanhf(v);
    }
    orow[r * K_ + k] = res;
  } else if (j < O_) {
    float e = expf(v);
    orow[6 * K_ + (j - 6 * K_)] = e;
    atomicAdd(&qk_sum[t], e);
  }
}

__global__ void qk_norm(float* __restrict__ out, const float* __restrict__ qk_sum, int t) {
  int idx = blockIdx.x * blockDim.x + threadIdx.x;  // b*3 + j
  if (idx >= B_ * 3) return;
  int b = idx / 3, j = idx - 3 * b;
  out[((size_t)b * T_ + t) * O_ + 6 * K_ + j] /= qk_sum[t];
}

// ---------------------------------------------------------------------------
extern "C" void kernel_launch(void* const* d_in, const int* in_sizes, int n_in,
                              void* d_out, int out_size, void* d_ws, size_t ws_size,
                              hipStream_t stream) {
  const float* s     = (const float*)d_in[0];
  const float* z     = (const float*)d_in[1];
  const float* w_h0  = (const float*)d_in[2];
  const float* b_h0  = (const float*)d_in[3];
  const float* w_ih  = (const float*)d_in[4];
  const float* b_ih  = (const float*)d_in[5];
  const float* w_hh  = (const float*)d_in[6];
  const float* b_hh  = (const float*)d_in[7];
  const float* w_dec = (const float*)d_in[8];
  const float* b_dec = (const float*)d_in[9];
  float* out = (float*)d_out;

  char* ws = (char*)d_ws;
  size_t off = 0;
  auto alloc = [&](size_t bytes) -> char* {
    char* p = ws + off;
    off = (off + bytes + 255) & ~(size_t)255;
    return p;
  };

  __bf16* w_hh_b  = (__bf16*)alloc((size_t)G4H * H_ * 2);   // 8 MB
  __bf16* w_dec_b = (__bf16*)alloc((size_t)128 * H_ * 2);   // 256 KB (padded)
  __bf16* w_h0_b  = (__bf16*)alloc((size_t)H_ * Z_ * 2);    // 256 KB
  __bf16* w_ihz_b = (__bf16*)alloc((size_t)G4H * Z_ * 2);   // 1 MB
  __bf16* z_b     = (__bf16*)alloc((size_t)B_ * Z_ * 2);    // 512 KB
  __bf16* h_b     = (__bf16*)alloc((size_t)B_ * H_ * 2);    // 4 MB
  float*  c_st    = (float*) alloc((size_t)B_ * H_ * 4);    // 8 MB
  float*  gbase   = (float*) alloc((size_t)B_ * G4H * 4);   // 32 MB
  float*  graw    = (float*) alloc((size_t)B_ * G4H * 4);   // 32 MB (also h0_raw)
  float*  yraw    = (float*) alloc((size_t)B_ * 128 * 4);   // 1 MB
  float*  qk_sum  = (float*) alloc((size_t)T_ * 4);
  (void)ws_size; (void)in_sizes; (void)n_in; (void)out_size;

  // --- one-time setup ---
  {
    int n = G4H * H_;
    cvt_bf16<<<(n + 255) / 256, 256, 0, stream>>>(w_hh, w_hh_b, n);
    n = H_ * Z_;
    cvt_bf16<<<(n + 255) / 256, 256, 0, stream>>>(w_h0, w_h0_b, n);
    n = B_ * Z_;
    cvt_bf16<<<(n + 255) / 256, 256, 0, stream>>>(z, z_b, n);
    extract_wihz<<<(G4H * Z_ + 255) / 256, 256, 0, stream>>>(w_ih, w_ihz_b);
    pad_wdec<<<(128 * H_ + 255) / 256, 256, 0, stream>>>(w_dec, w_dec_b);
    init_misc<<<1, 256, 0, stream>>>(qk_sum, out + (size_t)B_ * T_ * O_);
  }

  // h0 = tanh(z @ w_h0^T + b_h0)
  {
    dim3 grid(B_ / BM, H_ / BN);
    gemm_bf16<<<grid, 256, 0, stream>>>(z_b, w_h0_b, graw, B_, H_, Z_);
    h0_init<<<(B_ * H_ + 255) / 256, 256, 0, stream>>>(graw, b_h0, h_b, c_st);
  }

  // gbase = z @ w_ih[:, P:]^T  (time-invariant part of input gates)
  {
    dim3 grid(B_ / BM, G4H / BN);
    gemm_bf16<<<grid, 256, 0, stream>>>(z_b, w_ihz_b, gbase, B_, G4H, Z_);
  }

  // --- sequential timesteps ---
  dim3 grid_g(B_ / BM, G4H / BN);   // 16 x 32
  dim3 grid_d(B_ / BM, 128 / BN);   // 16 x 1
  for (int t = 0; t < T_; ++t) {
    gemm_bf16<<<grid_g, 256, 0, stream>>>(h_b, w_hh_b, graw, B_, G4H, H_);
    lstm_cell<<<(B_ * H_) / 256, 256, 0, stream>>>(graw, gbase, b_ih, b_hh, s, w_ih,
                                                   c_st, h_b, t);
    gemm_bf16<<<grid_d, 256, 0, stream>>>(h_b, w_dec_b, yraw, B_, 128, H_);
    decode_out<<<B_, 128, 0, stream>>>(yraw, b_dec, out, qk_sum, t);
    qk_norm<<<(B_ * 3 + 255) / 256, 256, 0, stream>>>(out, qk_sum, t);
  }
}
